// CPABTransform2D_22952305229996
// MI455X (gfx1250) — matrix operand order repaired and checked by reading, not compiled, not run on hardware
//
#include <hip/hip_runtime.h>
#include <hip/hip_bf16.h>

typedef __attribute__((ext_vector_type(16))) _Float16 v16h;
typedef __attribute__((ext_vector_type(8)))  _Float16 v8h;
typedef __attribute__((ext_vector_type(8)))  float    v8f;

#define NC_ 32
#define H_  64
#define D_  31

// d_ws layout (bytes)
#define OFF_W2H 0        // 4*2*32*16 half = 8192 B : W2 packed in WMMA A-operand order
#define OFF_W4H 8192     // 8192 B : W4 = B@W3 packed in WMMA A-operand order
#define OFF_UV  16384    // 2*32*16 float2 = 8192 B : fused layers 0+1 (u,v) in B-operand lane order
#define OFF_B2  24576    // 4*2*8 float = 256 B : b2 in C-init order
#define OFF_B4  24832    // 256 B : b4 = B@b3 in C-init order

// ---------------------------------------------------------------------------
// Setup: fold layers 0+1 into (u,v), fold basis into W4 = B@W3 / b4 = B@b3,
// and pre-pack weights into the exact WMMA A-operand per-lane layout so the
// main kernel does only contiguous 32B vector loads. One block, 64 threads.
// ---------------------------------------------------------------------------
__global__ void cpab_setup(const float* __restrict__ W0, const float* __restrict__ b0,
                           const float* __restrict__ W1, const float* __restrict__ b1,
                           const float* __restrict__ W2, const float* __restrict__ b2,
                           const float* __restrict__ W3, const float* __restrict__ b3,
                           const float* __restrict__ Bb, unsigned char* __restrict__ ws)
{
    __shared__ float su[H_], sv[H_], sb4[H_];
    __shared__ float sW4[H_ * H_];
    const int tid = threadIdx.x;

    if (tid < H_) {
        float u = 0.f, v = 0.f;
        for (int j = 0; j < H_; ++j) {
            float w = W1[tid * H_ + j];
            u += w * W0[j];          // W0 is (H,1)
            v += w * b0[j];
        }
        su[tid] = u;
        sv[tid] = v + b1[tid];
        float bb = 0.f;
        for (int d = 0; d < D_; ++d) bb += Bb[tid * D_ + d] * b3[d];
        sb4[tid] = bb;
        for (int k = 0; k < H_; ++k) {
            float s = 0.f;
            for (int d = 0; d < D_; ++d) s += Bb[tid * D_ + d] * W3[d * H_ + k];
            sW4[tid * H_ + k] = s;   // W4 = B @ W3  (64x64)
        }
    }
    __syncthreads();

    // A-operand pack (contiguous k, same order as B): elems 0..15 -> K=base..base+15,
    // base = c*32 + (lane>=16 ? 16 : 0). Row M = t*16 + (lane&15).
    _Float16* W2h = (_Float16*)(ws + OFF_W2H);
    _Float16* W4h = (_Float16*)(ws + OFF_W4H);
    for (int idx = tid; idx < 4 * 2 * 32 * 16; idx += blockDim.x) {
        int e = idx & 15, L = (idx >> 4) & 31, c = (idx >> 9) & 1, t = idx >> 10;
        int k = c * 32 + ((L >> 4) << 4) + e;
        int h = t * 16 + (L & 15);
        W2h[idx] = (_Float16)W2[h * H_ + k];
        W4h[idx] = (_Float16)sW4[h * H_ + k];
    }
    // B-operand pack for (u,v): lane<16 holds K=c*32+0..15, lane>=16 K=c*32+16..31.
    float2* uvp = (float2*)(ws + OFF_UV);
    for (int idx = tid; idx < 2 * 32 * 16; idx += blockDim.x) {
        int e = idx & 15, L = (idx >> 4) & 31, c = idx >> 9;
        int k = c * 32 + ((L >> 4) << 4) + e;
        uvp[idx] = make_float2(su[k], sv[k]);
    }
    // C-init bias pack: element j of v8f for (tile t, lane-half hf) = bias[t*16 + hf*8 + j]
    float* b2p = (float*)(ws + OFF_B2);
    float* b4p = (float*)(ws + OFF_B4);
    if (tid < 64) {
        int j = tid & 7, hf = (tid >> 3) & 1, t = tid >> 4;
        int h = t * 16 + hf * 8 + j;
        b2p[tid] = b2[h];
        b4p[tid] = sb4[h];
    }
}

// piecewise-affine velocity lookup: (a*x + b, a) for cell c = clip(floor(x*32),0,31)
__device__ __forceinline__ void cpab_vel(const float* __restrict__ Arow, float q,
                                         float& vout, float& aout)
{
    int c = (int)floorf(q * (float)NC_);
    c = c < 0 ? 0 : (c > NC_ - 1 ? NC_ - 1 : c);
    float2 ab = *(const float2*)(Arow + 2 * c);   // (Aa[c], Ab[c]) interleaved
    aout = ab.x;
    vout = ab.x * q + ab.y;
}

// ---------------------------------------------------------------------------
// Main kernel: one wave = 32 points. Two 64x64 f16 WMMA GEMMs (32 v_wmma per
// wave), LDS bounce for the layer2->layer3 operand re-layout, LDS-resident A
// coefficients for the RK4 cell lookups.
// ---------------------------------------------------------------------------
__global__ __launch_bounds__(256) void cpab_main(const float* __restrict__ x,
                                                 const unsigned char* __restrict__ ws,
                                                 float* __restrict__ out, int N)
{
    __shared__ __align__(16) char smem[8 * 8192];   // 8KB per wave (h2 f16 overlaid with A f32)
    const int lane = threadIdx.x & 31;
    const int warp = threadIdx.x >> 5;
    const int lh   = lane >> 4;       // lane half (0: lanes 0-15, 1: lanes 16-31)
    const int lm   = lane & 15;
    char* wsm = smem + warp * 8192;

    const int ntiles = (N + 31) >> 5;
    const int wid = blockIdx.x * (blockDim.x >> 5) + warp;
    if (wid >= ntiles) return;        // wave-uniform: EXEC stays all-ones for WMMA
    const int p0 = wid << 5;

    // ---- load pre-packed constant operands ----
    const v16h* W2p = (const v16h*)(ws + OFF_W2H);
    const v16h* W4p = (const v16h*)(ws + OFF_W4H);
    v16h w2[4][2], w4[4][2];
#pragma unroll
    for (int t = 0; t < 4; ++t)
#pragma unroll
        for (int c = 0; c < 2; ++c) {
            w2[t][c] = W2p[(t * 2 + c) * 32 + lane];
            w4[t][c] = W4p[(t * 2 + c) * 32 + lane];
        }
    v8f bias2[4], bias4[4];
#pragma unroll
    for (int t = 0; t < 4; ++t) {
        bias2[t] = ((const v8f*)(ws + OFF_B2))[t * 2 + lh];
        bias4[t] = ((const v8f*)(ws + OFF_B4))[t * 2 + lh];
    }

    // ---- fused layers 0+1 (affine in x1), built directly as WMMA B operands ----
    const int ptA = min(p0 + lm, N - 1);
    const int ptB = min(p0 + 16 + lm, N - 1);
    const float x1A = x[2 * ptA + 1];
    const float x1B = x[2 * ptB + 1];
    const float2* UV = (const float2*)(ws + OFF_UV);
    v16h hA[2], hB[2];
#pragma unroll
    for (int c = 0; c < 2; ++c)
#pragma unroll
        for (int e = 0; e < 16; ++e) {
            float2 w = UV[(c * 32 + lane) * 16 + e];
            hA[c][e] = (_Float16)fmaxf(0.f, fmaf(w.x, x1A, w.y));
            hB[c][e] = (_Float16)fmaxf(0.f, fmaf(w.x, x1B, w.y));
        }

    // ---- layer 2: D[h_out, m] = W2 x relu(h1)^T + b2 ----
    v8f a2A[4], a2B[4];
#pragma unroll
    for (int t = 0; t < 4; ++t) {
        a2A[t] = __builtin_amdgcn_wmma_f32_16x16x32_f16(false, w2[t][0], false, hA[0], (short)0, bias2[t], false, false);
        a2A[t] = __builtin_amdgcn_wmma_f32_16x16x32_f16(false, w2[t][1], false, hA[1], (short)0, a2A[t],   false, false);
        a2B[t] = __builtin_amdgcn_wmma_f32_16x16x32_f16(false, w2[t][0], false, hB[0], (short)0, bias2[t], false, false);
        a2B[t] = __builtin_amdgcn_wmma_f32_16x16x32_f16(false, w2[t][1], false, hB[1], (short)0, a2B[t],   false, false);
    }

    // relu + f16, stage h2 to LDS as [m (32 rows) x k (64)] for B-operand re-layout
    _Float16* h2s = (_Float16*)wsm;
#pragma unroll
    for (int t = 0; t < 4; ++t) {
        v8h pa, pb;
#pragma unroll
        for (int j = 0; j < 8; ++j) {
            pa[j] = (_Float16)fmaxf(0.f, a2A[t][j]);   // row h = t*16 + lh*8 + j, col m = lm
            pb[j] = (_Float16)fmaxf(0.f, a2B[t][j]);   // col m = 16 + lm
        }
        *(v8h*)(h2s + lm * 64        + t * 16 + lh * 8) = pa;
        *(v8h*)(h2s + (16 + lm) * 64 + t * 16 + lh * 8) = pb;
    }
    asm volatile("s_wait_dscnt 0" ::: "memory");

    // ---- layer 3 B operands: contiguous 32B v16h per lane from LDS ----
    v16h gA[2], gB[2];
#pragma unroll
    for (int c = 0; c < 2; ++c) {
        gA[c] = *(const v16h*)(h2s + lm * 64        + c * 32 + lh * 16);
        gB[c] = *(const v16h*)(h2s + (16 + lm) * 64 + c * 32 + lh * 16);
    }
    asm volatile("s_wait_dscnt 0" ::: "memory");   // fence before overlaying region with f32 A

    // ---- layer 3: A_flat[h, m] = W4 x h2^T + b4 ----
    v8f a3A[4], a3B[4];
#pragma unroll
    for (int t = 0; t < 4; ++t) {
        a3A[t] = __builtin_amdgcn_wmma_f32_16x16x32_f16(false, w4[t][0], false, gA[0], (short)0, bias4[t], false, false);
        a3A[t] = __builtin_amdgcn_wmma_f32_16x16x32_f16(false, w4[t][1], false, gA[1], (short)0, a3A[t],   false, false);
        a3B[t] = __builtin_amdgcn_wmma_f32_16x16x32_f16(false, w4[t][0], false, gB[0], (short)0, bias4[t], false, false);
        a3B[t] = __builtin_amdgcn_wmma_f32_16x16x32_f16(false, w4[t][1], false, gB[1], (short)0, a3B[t],   false, false);
    }

    // ---- store A coefficients (f32) to LDS: row m (32) x h (64); h=2c -> Aa[c], h=2c+1 -> Ab[c]
    float* As = (float*)wsm;
#pragma unroll
    for (int t = 0; t < 4; ++t) {
        float* dA = As + lm * 64        + t * 16 + lh * 8;
        float* dB = As + (16 + lm) * 64 + t * 16 + lh * 8;
        *(float4*)(dA)     = make_float4(a3A[t][0], a3A[t][1], a3A[t][2], a3A[t][3]);
        *(float4*)(dA + 4) = make_float4(a3A[t][4], a3A[t][5], a3A[t][6], a3A[t][7]);
        *(float4*)(dB)     = make_float4(a3B[t][0], a3B[t][1], a3B[t][2], a3B[t][3]);
        *(float4*)(dB + 4) = make_float4(a3B[t][4], a3B[t][5], a3B[t][6], a3B[t][7]);
    }
    asm volatile("s_wait_dscnt 0" ::: "memory");

    // ---- RK4 integration, one point per lane ----
    const int pt = p0 + lane;
    if (pt < N) {
        const float xin0 = x[2 * pt];
        const float xin1 = x[2 * pt + 1];
        const float* Arow = (const float*)wsm + lane * 64;
        float xp = xin0, lj = 0.f;
        const float dt = 1.f / 16.f;
#pragma unroll 4
        for (int s = 0; s < 16; ++s) {
            float k1v, k1a, k2v, k2a, k3v, k3a, k4v, k4a;
            cpab_vel(Arow, xp,                  k1v, k1a);
            cpab_vel(Arow, xp + 0.5f * dt * k1v, k2v, k2a);
            cpab_vel(Arow, xp + 0.5f * dt * k2v, k3v, k3a);
            cpab_vel(Arow, xp + dt * k3v,        k4v, k4a);
            xp = xp + (dt / 6.f) * (k1v + 2.f * k2v + 2.f * k3v + k4v);
            lj = lj + (dt / 6.f) * (k1a + 2.f * k2a + 2.f * k3a + k4a);
        }
        out[2 * pt]         = xp;       // z[:,0]
        out[2 * pt + 1]     = xin1;     // z[:,1]
        out[2 * N + 2 * pt]     = lj;   // log_dz_dx[:,0]
        out[2 * N + 2 * pt + 1] = 0.f;  // log_dz_dx[:,1]
    }
}

extern "C" void kernel_launch(void* const* d_in, const int* in_sizes, int n_in,
                              void* d_out, int out_size, void* d_ws, size_t ws_size,
                              hipStream_t stream)
{
    const float* x  = (const float*)d_in[0];
    const float* W0 = (const float*)d_in[1];
    const float* b0 = (const float*)d_in[2];
    const float* W1 = (const float*)d_in[3];
    const float* b1 = (const float*)d_in[4];
    const float* W2 = (const float*)d_in[5];
    const float* b2 = (const float*)d_in[6];
    const float* W3 = (const float*)d_in[7];
    const float* b3 = (const float*)d_in[8];
    const float* Bb = (const float*)d_in[9];

    const int N = in_sizes[0] / 2;

    cpab_setup<<<1, 64, 0, stream>>>(W0, b0, W1, b1, W2, b2, W3, b3, Bb,
                                     (unsigned char*)d_ws);

    const int ntiles = (N + 31) / 32;   // one 32-point tile per wave
    const int blocks = (ntiles + 7) / 8;
    cpab_main<<<blocks, 256, 0, stream>>>(x, (const unsigned char*)d_ws,
                                          (float*)d_out, N);
}
